// StructureEncoder_62766652064155
// MI455X (gfx1250) — compile-verified, matrix operand order checked by reading, not run on hardware
//
#include <hip/hip_runtime.h>
#include <hip/hip_bf16.h>

#define N_NODES  50000
#define N_EDGES  1600000
#define HID      128
#define N_LAYERS 3
#define N_GRAPHS 64
#define LN_EPS   1e-5f

typedef __attribute__((ext_vector_type(16))) __bf16 v16bf;
typedef __attribute__((ext_vector_type(8)))  __bf16 v8bf;
typedef __attribute__((ext_vector_type(8)))  float  v8f;

// ---------------------------------------------------------------------------
// Degree / inverse degree
// ---------------------------------------------------------------------------
__global__ void deg_kernel(const int* __restrict__ dst, float* __restrict__ deg) {
    int e = blockIdx.x * blockDim.x + threadIdx.x;
    if (e < N_EDGES) atomicAdd(&deg[dst[e]], 1.0f);
}

__global__ void invdeg_kernel(float* __restrict__ deg) {
    int i = blockIdx.x * blockDim.x + threadIdx.x;
    if (i < N_NODES) deg[i] = 1.0f / fmaxf(deg[i], 1.0f);
}

// ---------------------------------------------------------------------------
// Edge scatter: agg[dst] += x[src]   (32 lanes per edge, float4 per lane)
// ---------------------------------------------------------------------------
__global__ __launch_bounds__(256) void scatter_kernel(
    const float* __restrict__ x, const int* __restrict__ src,
    const int* __restrict__ dst, float* __restrict__ agg) {
    int gid = blockIdx.x * 256 + threadIdx.x;
    int e = gid >> 5;
    if (e >= N_EDGES) return;
    int q = (gid & 31) * 4;
    int s = src[e], d = dst[e];
    const float4 v = *(const float4*)(x + (size_t)s * HID + q);
    float* p = agg + (size_t)d * HID + q;
    atomicAdd(p + 0, v.x);
    atomicAdd(p + 1, v.y);
    atomicAdd(p + 2, v.z);
    atomicAdd(p + 3, v.w);
}

// ---------------------------------------------------------------------------
// Weight swizzle: Wcat = [Wn; Wr] (256x128 f32) -> bf16 in per-lane WMMA
// B-fragment order: Wswz[((nt*8 + kt)*32 + lane)*16 + j]
//   n = nt*16 + (lane&15); kb = kt*32 + (lane>>4)*8
//   j<8 : K = kb + j ;  j>=8 : K = kb + 16 + (j-8)
// ---------------------------------------------------------------------------
__global__ __launch_bounds__(256) void swizzle_w_kernel(
    const float* __restrict__ Wn, const float* __restrict__ Wr,
    __bf16* __restrict__ Wswz) {
    int idx = blockIdx.x * 256 + threadIdx.x;   // 0 .. 32767
    int j    = idx & 15;
    int lane = (idx >> 4) & 31;
    int kt   = (idx >> 9) & 7;
    int nt   = idx >> 12;
    int n  = nt * 16 + (lane & 15);
    int kb = kt * 32 + (lane >> 4) * 8;
    int k  = kb + ((j < 8) ? j : (8 + j));      // j>=8: kb + 16 + (j-8)
    float v = (k < HID) ? Wn[k * HID + n] : Wr[(k - HID) * HID + n];
    Wswz[idx] = (__bf16)v;
}

// ---------------------------------------------------------------------------
// Fused:  h = [agg*inv_deg | x] @ [Wn; Wr] + bn ; LayerNorm ; ReLU -> xout
// Block = 256 threads = 8 waves; block tile = 16 rows x 128 cols;
// wave w owns cols [16w, 16w+16); K = 256 in 8 WMMA steps of 32.
// ---------------------------------------------------------------------------
__global__ __launch_bounds__(256) void gemm_ln_relu_kernel(
    const float* __restrict__ agg, const float* __restrict__ inv_deg,
    const float* __restrict__ xin, const __bf16* __restrict__ Wswz,
    const float* __restrict__ bn, const float* __restrict__ gamma,
    const float* __restrict__ beta, float* __restrict__ xout) {
    const int tid  = threadIdx.x;
    const int wave = tid >> 5;
    const int lane = tid & 31;
    const int row0 = blockIdx.x * 16;

    __shared__ __bf16 sA[16][264];   // 16 rows x 256 K (bf16), padded
    __shared__ float  sH[16][132];   // h tile (f32), padded

    // --- stage A tile: rows = nodes, K 0..127 = agg*inv_deg, 128..255 = x ---
    {
        const int r    = tid >> 4;          // 0..15
        const int c0   = (tid & 15) * 16;   // 0,16,...,240
        const int node = row0 + r;
        const float scale = inv_deg[node];
        const float* arow = agg + (size_t)node * HID;
        const float* xrow = xin + (size_t)node * HID;
        #pragma unroll
        for (int j = 0; j < 16; ++j) {
            int k = c0 + j;
            float v = (k < HID) ? (arow[k] * scale) : xrow[k - HID];
            sA[r][k] = (__bf16)v;
        }
    }
    __syncthreads();

    const int ncol  = wave * 16 + (lane & 15);
    const int khalf = (lane >> 4) * 8;

    v8f acc;
    {
        float b = bn[ncol];
        #pragma unroll
        for (int i = 0; i < 8; ++i) acc[i] = b;
    }

    #pragma unroll
    for (int kt = 0; kt < 8; ++kt) {
        const int kb = kt * 32 + khalf;
        const int r  = lane & 15;
        v8bf lo = *(const v8bf*)&sA[r][kb];
        v8bf hi = *(const v8bf*)&sA[r][kb + 16];
        v16bf afrag = __builtin_shufflevector(lo, hi,
            0, 1, 2, 3, 4, 5, 6, 7, 8, 9, 10, 11, 12, 13, 14, 15);
        v16bf bfrag = *(const v16bf*)(Wswz + (((wave * 8 + kt) * 32 + lane) << 4));
        acc = __builtin_amdgcn_wmma_f32_16x16x32_bf16(
            false, afrag, false, bfrag, (short)0, acc, false, false);
    }

    // --- spill tile to LDS: acc[i] is (row = i + 8*(lane>>4), col = ncol) ---
    {
        const int mhi = (lane >> 4) * 8;
        #pragma unroll
        for (int i = 0; i < 8; ++i) sH[mhi + i][ncol] = acc[i];
    }
    __syncthreads();

    // --- LayerNorm + ReLU: 16 threads per row, 8 cols each ---
    {
        const int r   = tid >> 4;
        const int sub = tid & 15;
        float s = 0.f, s2 = 0.f;
        #pragma unroll
        for (int j = 0; j < 8; ++j) {
            float v = sH[r][sub + 16 * j];
            s += v; s2 += v * v;
        }
        #pragma unroll
        for (int m = 1; m < 16; m <<= 1) {
            s  += __shfl_xor(s,  m, 32);
            s2 += __shfl_xor(s2, m, 32);
        }
        const float mu   = s * (1.0f / HID);
        const float var  = s2 * (1.0f / HID) - mu * mu;
        const float rstd = rsqrtf(var + LN_EPS);
        const int node = row0 + r;
        float* orow = xout + (size_t)node * HID;
        #pragma unroll
        for (int j = 0; j < 8; ++j) {
            int c = sub + 16 * j;
            float v = (sH[r][c] - mu) * rstd * gamma[c] + beta[c];
            orow[c] = fmaxf(v, 0.0f);
        }
    }
}

// ---------------------------------------------------------------------------
// Graph pooling
// ---------------------------------------------------------------------------
__global__ void pool_cnt_kernel(const int* __restrict__ batch, float* __restrict__ cnt) {
    int i = blockIdx.x * blockDim.x + threadIdx.x;
    if (i < N_NODES) atomicAdd(&cnt[batch[i]], 1.0f);
}

__global__ void pool_sum_kernel(const float* __restrict__ node_emb,
                                const int* __restrict__ batch,
                                float* __restrict__ gacc) {
    int gid = blockIdx.x * blockDim.x + threadIdx.x;
    int node = gid >> 7;
    if (node >= N_NODES) return;
    int c = gid & 127;
    atomicAdd(&gacc[batch[node] * HID + c], node_emb[(size_t)node * HID + c]);
}

__global__ void pool_final_kernel(const float* __restrict__ gacc,
                                  const float* __restrict__ cnt,
                                  float* __restrict__ gout) {
    int i = blockIdx.x * blockDim.x + threadIdx.x;
    if (i < N_GRAPHS * HID) {
        int g = i >> 7;
        gout[i] = gacc[i] / fmaxf(cnt[g], 1.0f);
    }
}

// ---------------------------------------------------------------------------
extern "C" void kernel_launch(void* const* d_in, const int* in_sizes, int n_in,
                              void* d_out, int out_size, void* d_ws, size_t ws_size,
                              hipStream_t stream) {
    const float* x     = (const float*)d_in[0];
    const float* Wn    = (const float*)d_in[1];
    const float* bn    = (const float*)d_in[2];
    const float* Wr    = (const float*)d_in[3];
    const float* gamma = (const float*)d_in[4];
    const float* beta  = (const float*)d_in[5];
    const int*   ei    = (const int*)d_in[6];
    const int*   batch = (const int*)d_in[7];
    const int* src = ei;
    const int* dst = ei + N_EDGES;

    float* gout     = (float*)d_out;                 // 64 x 128
    float* node_emb = gout + N_GRAPHS * HID;         // 50000 x 128

    // workspace layout (floats)
    float*  agg  = (float*)d_ws;                     // N_NODES*HID
    float*  xbuf = agg + (size_t)N_NODES * HID;      // N_NODES*HID
    float*  deg  = xbuf + (size_t)N_NODES * HID;     // N_NODES (-> inv_deg)
    float*  cnt  = deg + N_NODES;                    // N_GRAPHS
    float*  gacc = cnt + N_GRAPHS;                   // N_GRAPHS*HID
    __bf16* Wswz = (__bf16*)(gacc + N_GRAPHS * HID); // 256*128 bf16

    // zero deg + cnt + gacc (contiguous)
    hipMemsetAsync(deg, 0, (size_t)(N_NODES + N_GRAPHS + N_GRAPHS * HID) * sizeof(float), stream);

    deg_kernel<<<(N_EDGES + 255) / 256, 256, 0, stream>>>(dst, deg);
    invdeg_kernel<<<(N_NODES + 255) / 256, 256, 0, stream>>>(deg);

    const float* xin = x;
    for (int l = 0; l < N_LAYERS; ++l) {
        float* xout = (l == N_LAYERS - 1) ? node_emb : xbuf;
        swizzle_w_kernel<<<128, 256, 0, stream>>>(Wn + (size_t)l * HID * HID,
                                                  Wr + (size_t)l * HID * HID, Wswz);
        hipMemsetAsync(agg, 0, (size_t)N_NODES * HID * sizeof(float), stream);
        scatter_kernel<<<(N_EDGES * 32 + 255) / 256, 256, 0, stream>>>(xin, src, dst, agg);
        gemm_ln_relu_kernel<<<N_NODES / 16, 256, 0, stream>>>(
            agg, deg, xin, Wswz, bn + l * HID, gamma + l * HID, beta + l * HID, xout);
        xin = xout;
    }

    pool_cnt_kernel<<<(N_NODES + 255) / 256, 256, 0, stream>>>(batch, cnt);
    pool_sum_kernel<<<((size_t)N_NODES * HID + 255) / 256, 256, 0, stream>>>(node_emb, batch, gacc);
    pool_final_kernel<<<(N_GRAPHS * HID + 255) / 256, 256, 0, stream>>>(gacc, cnt, gout);
}
